// GraphAttentionLayer_87058987090398
// MI455X (gfx1250) — matrix operand order repaired and checked by reading, not compiled, not run on hardware
//
#include <hip/hip_runtime.h>
#include <hip/hip_bf16.h>

typedef __attribute__((ext_vector_type(16))) __bf16 v16bf;
typedef __attribute__((ext_vector_type(8)))  __bf16 bf16x8;
typedef __attribute__((ext_vector_type(8)))  float  v8f;
typedef __attribute__((ext_vector_type(4)))  float  f32x4;

#define N_NODES 4096
#define IN_F    256
#define OUT_F   64
#define HEADS   8
#define LRELU_ALPHA 0.2f
#define NEG_INF (-__builtin_inff())

// ---------------- prep kernels ----------------

__global__ void cvt_h_kernel(const float* __restrict__ h, __bf16* __restrict__ hb, int n) {
    int i = blockIdx.x * blockDim.x + threadIdx.x;
    if (i < n) hb[i] = (__bf16)h[i];
}

// W (H, IN, OUT) f32 -> WT (H, OUT, IN) bf16
__global__ void cvt_WT_kernel(const float* __restrict__ W, __bf16* __restrict__ WT) {
    int idx = blockIdx.x * blockDim.x + threadIdx.x;   // H*IN*OUT
    int hh = idx / (IN_F * OUT_F);
    int r  = idx % (IN_F * OUT_F);
    int i  = r / OUT_F;
    int o  = r % OUT_F;
    WT[(hh * OUT_F + o) * IN_F + i] = (__bf16)W[idx];
}

// ---------------- Wh = h @ W[h], stored transposed: WhT (H, OUT, N) bf16 ----------------
// One wave per 16(n) x 16(o) output tile; K-loop over IN in steps of 32.
__global__ void __launch_bounds__(256) wh_gemm_kernel(const __bf16* __restrict__ hb,
                                                      const __bf16* __restrict__ WT,
                                                      __bf16* __restrict__ WhT) {
    const int wave = threadIdx.x >> 5;
    const int lane = threadIdx.x & 31;
    const int tile = blockIdx.x * 8 + wave;
    const int TPH  = (N_NODES / 16) * (OUT_F / 16);     // tiles per head
    const int hh   = tile / TPH;
    const int rem  = tile % TPH;
    const int nt   = rem / (OUT_F / 16);
    const int ot   = rem % (OUT_F / 16);

    const int r  = lane & 15;
    const int hi = lane >> 4;
    const int row  = nt * 16 + r;           // A-matrix row (node)
    const int ocol = ot * 16 + r;           // B-matrix column (out feature)

    v8f acc = {};
    const __bf16* Arow = hb + (size_t)row * IN_F;
    const __bf16* Bcol = WT + ((size_t)hh * OUT_F + ocol) * IN_F;

    #pragma unroll
    for (int kb = 0; kb < IN_F; kb += 32) {
        union { v16bf v; bf16x8 h2[2]; } A, B;
        // A-layout: lane(row) holds K = hi*16 .. hi*16+15 (contiguous)
        A.h2[0] = *(const bf16x8*)(Arow + kb + hi * 16);
        A.h2[1] = *(const bf16x8*)(Arow + kb + hi * 16 + 8);
        // B-layout: lane(col) holds K = hi*16 .. hi*16+15 (contiguous)
        B.h2[0] = *(const bf16x8*)(Bcol + kb + hi * 16);
        B.h2[1] = *(const bf16x8*)(Bcol + kb + hi * 16 + 8);
        acc = __builtin_amdgcn_wmma_f32_16x16x32_bf16(false, A.v, false, B.v,
                                                      (short)0, acc, false, false);
    }

    // D layout: VGPR r2 -> row M = r2 + 8*hi, col = lane&15.
    // Write WhT[h][ocol][nt*16 + 8*hi + r2] : 8 consecutive n -> one 16B store.
    bf16x8 outv;
    #pragma unroll
    for (int r2 = 0; r2 < 8; ++r2) outv[r2] = (__bf16)acc[r2];
    *(bf16x8*)(WhT + ((size_t)hh * OUT_F + ocol) * N_NODES + nt * 16 + hi * 8) = outv;
}

// ---------------- src/dst projection vectors ----------------
__global__ void src_dst_kernel(const __bf16* __restrict__ WhT, const float* __restrict__ a,
                               float* __restrict__ srcv, float* __restrict__ dstv) {
    int idx = blockIdx.x * blockDim.x + threadIdx.x;   // H*N
    int hh = idx / N_NODES;
    int n  = idx % N_NODES;
    float s = 0.f, d = 0.f;
    #pragma unroll 8
    for (int o = 0; o < OUT_F; ++o) {
        float w = (float)WhT[((size_t)hh * OUT_F + o) * N_NODES + n];
        s += w * a[hh * 2 * OUT_F + o];
        d += w * a[hh * 2 * OUT_F + OUT_F + o];
    }
    srcv[idx] = s;
    dstv[idx] = d;
}

// ---------------- per-head max of dst (static softmax shift) ----------------
__global__ void __launch_bounds__(256) dmax_kernel(const float* __restrict__ dstv,
                                                   float* __restrict__ dmax) {
    const int hh = blockIdx.x;
    const int t  = threadIdx.x;
    float mx = NEG_INF;
    for (int n = t; n < N_NODES; n += 256) mx = fmaxf(mx, dstv[hh * N_NODES + n]);
    #pragma unroll
    for (int off = 16; off >= 1; off >>= 1) mx = fmaxf(mx, __shfl_xor(mx, off, 32));
    __shared__ float red[8];
    if ((t & 31) == 0) red[t >> 5] = mx;
    __syncthreads();
    if (t == 0) {
        float m = red[0];
        #pragma unroll
        for (int i = 1; i < 8; ++i) m = fmaxf(m, red[i]);
        dmax[hh] = m;
    }
}

// ---------------- fused masked-softmax attention + aggregation ----------------
// Block = 8 waves (one head each) on one 16-row i-tile.
// Static softmax shift: leaky_relu is monotone, so
//   max_j leaky_relu(src_i + dst_j) <= leaky_relu(src_i + max_j dst_j) = m_i  (exact bound)
// -> single pass, no running max, no accumulator rescale.
// P (bf16) @ WhT via 4x v_wmma_f32_16x16x32_bf16 per 32-column step.
__global__ void __launch_bounds__(256) gat_attn_kernel(const float* __restrict__ adj,
                                                       const __bf16* __restrict__ WhT,
                                                       const float* __restrict__ srcv,
                                                       const float* __restrict__ dstv,
                                                       const float* __restrict__ dmax,
                                                       float* __restrict__ out) {
    const int hh   = threadIdx.x >> 5;   // wave id == head
    const int lane = threadIdx.x & 31;
    const int it   = blockIdx.x;         // i-tile (16 rows)
    const int r    = lane & 15;
    const int hi   = lane >> 4;
    const int row  = it * 16 + r;

    const float src_i = srcv[hh * N_NODES + row];
    // per-row stability shift (uniform for both half-lanes of a row)
    float mrow = src_i + dmax[hh];
    mrow = fmaxf(mrow, LRELU_ALPHA * mrow);  // leaky_relu, monotone bound

    const float* __restrict__ adjrow = adj + (size_t)row * N_NODES;
    const float* __restrict__ dstp   = dstv + (size_t)hh * N_NODES;
    const __bf16* __restrict__ Wb    = WhT + (size_t)hh * OUT_F * N_NODES;

    float l = 0.f;   // per-lane partial row sum (combined across lane pair at end)
    v8f acc0 = {}, acc1 = {}, acc2 = {}, acc3 = {};

    for (int jb = 0; jb < N_NODES; jb += 32) {
        if (jb + 32 < N_NODES) {
            __builtin_prefetch(adjrow + jb + 32 + hi * 8, 0, 1);
            __builtin_prefetch(adjrow + jb + 48 + hi * 8, 0, 1);
        }
        const int j0 = jb + hi * 8;        // k = 0..7  -> columns j0..j0+7
        const int j1 = jb + 16 + hi * 8;   // k = 8..15 -> columns j1..j1+7

        f32x4 a0 = *(const f32x4*)(adjrow + j0);
        f32x4 a1 = *(const f32x4*)(adjrow + j0 + 4);
        f32x4 a2 = *(const f32x4*)(adjrow + j1);
        f32x4 a3 = *(const f32x4*)(adjrow + j1 + 4);
        f32x4 d0 = *(const f32x4*)(dstp + j0);
        f32x4 d1 = *(const f32x4*)(dstp + j0 + 4);
        f32x4 d2 = *(const f32x4*)(dstp + j1);
        f32x4 d3 = *(const f32x4*)(dstp + j1 + 4);

        float adjv[16], dv[16];
        #pragma unroll
        for (int k = 0; k < 4; ++k) {
            adjv[k]      = a0[k]; dv[k]      = d0[k];
            adjv[k + 4]  = a1[k]; dv[k + 4]  = d1[k];
            adjv[k + 8]  = a2[k]; dv[k + 8]  = d2[k];
            adjv[k + 12] = a3[k]; dv[k + 12] = d3[k];
        }

        // p = adj * exp(leaky_relu(src_i + dst_j) - m); adj is exactly 0.0/1.0
        union { v16bf v; __bf16 e16[16]; } A;
        #pragma unroll
        for (int k = 0; k < 16; ++k) {
            float x = src_i + dv[k];
            x = fmaxf(x, LRELU_ALPHA * x);              // leaky_relu
            float pk = __expf(x - mrow) * adjv[k];      // mask by multiply
            l += pk;
            A.e16[k] = (__bf16)pk;
        }

        // B = WhT tiles: lane(col o) holds j = {hi*8..hi*8+7} U {16+hi*8..16+hi*8+7} -> two 16B loads each
        const __bf16* bp0 = Wb + (size_t)(0 * 16 + r) * N_NODES + jb + hi * 8;
        const __bf16* bp1 = Wb + (size_t)(1 * 16 + r) * N_NODES + jb + hi * 8;
        const __bf16* bp2 = Wb + (size_t)(2 * 16 + r) * N_NODES + jb + hi * 8;
        const __bf16* bp3 = Wb + (size_t)(3 * 16 + r) * N_NODES + jb + hi * 8;
        union { v16bf v; bf16x8 h2[2]; } B0, B1, B2, B3;
        B0.h2[0] = *(const bf16x8*)bp0; B0.h2[1] = *(const bf16x8*)(bp0 + 16);
        B1.h2[0] = *(const bf16x8*)bp1; B1.h2[1] = *(const bf16x8*)(bp1 + 16);
        B2.h2[0] = *(const bf16x8*)bp2; B2.h2[1] = *(const bf16x8*)(bp2 + 16);
        B3.h2[0] = *(const bf16x8*)bp3; B3.h2[1] = *(const bf16x8*)(bp3 + 16);

        acc0 = __builtin_amdgcn_wmma_f32_16x16x32_bf16(false, A.v, false, B0.v, (short)0, acc0, false, false);
        acc1 = __builtin_amdgcn_wmma_f32_16x16x32_bf16(false, A.v, false, B1.v, (short)0, acc1, false, false);
        acc2 = __builtin_amdgcn_wmma_f32_16x16x32_bf16(false, A.v, false, B2.v, (short)0, acc2, false, false);
        acc3 = __builtin_amdgcn_wmma_f32_16x16x32_bf16(false, A.v, false, B3.v, (short)0, acc3, false, false);
    }

    // combine partial row sums across the lane pair (rows split over lanes L, L+16)
    l += __shfl_xor(l, 16, 32);

    // finalize: divide by l (0 if row fully masked -> matches nan_to_num), ELU, store
    float linv[8];
    #pragma unroll
    for (int r2 = 0; r2 < 8; ++r2) {
        float lr = __shfl(l, hi * 8 + r2, 32);   // lane p holds total for row p
        linv[r2] = (lr > 0.f) ? (1.f / lr) : 0.f;
    }
    #pragma unroll
    for (int r2 = 0; r2 < 8; ++r2) {
        const int n = it * 16 + r2 + 8 * hi;
        float* __restrict__ orow = out + (size_t)n * (HEADS * OUT_F) + hh * OUT_F;
        float v0 = acc0[r2] * linv[r2];
        float v1 = acc1[r2] * linv[r2];
        float v2 = acc2[r2] * linv[r2];
        float v3 = acc3[r2] * linv[r2];
        v0 = (v0 > 0.f) ? v0 : (__expf(v0) - 1.f);
        v1 = (v1 > 0.f) ? v1 : (__expf(v1) - 1.f);
        v2 = (v2 > 0.f) ? v2 : (__expf(v2) - 1.f);
        v3 = (v3 > 0.f) ? v3 : (__expf(v3) - 1.f);
        orow[0 * 16 + r] = v0;
        orow[1 * 16 + r] = v1;
        orow[2 * 16 + r] = v2;
        orow[3 * 16 + r] = v3;
    }
}

// ---------------- launch ----------------

extern "C" void kernel_launch(void* const* d_in, const int* in_sizes, int n_in,
                              void* d_out, int out_size, void* d_ws, size_t ws_size,
                              hipStream_t stream) {
    const float* h_f   = (const float*)d_in[0];   // (N, IN)
    const float* adj_f = (const float*)d_in[1];   // (N, N)
    const float* W_f   = (const float*)d_in[2];   // (H, IN, OUT)
    const float* a_f   = (const float*)d_in[3];   // (H, 2*OUT)
    float* out_f = (float*)d_out;                 // (N, H*OUT)

    // workspace layout (256B aligned)
    char* ws = (char*)d_ws;
    size_t off = 0;
    auto alloc = [&](size_t bytes) { void* p = ws + off; off = (off + bytes + 255) & ~(size_t)255; return p; };
    __bf16* hb   = (__bf16*)alloc((size_t)N_NODES * IN_F * sizeof(__bf16));
    __bf16* WT   = (__bf16*)alloc((size_t)HEADS * OUT_F * IN_F * sizeof(__bf16));
    __bf16* WhT  = (__bf16*)alloc((size_t)HEADS * OUT_F * N_NODES * sizeof(__bf16));
    float*  srcv = (float*) alloc((size_t)HEADS * N_NODES * sizeof(float));
    float*  dstv = (float*) alloc((size_t)HEADS * N_NODES * sizeof(float));
    float*  dmx  = (float*) alloc((size_t)HEADS * sizeof(float));
    (void)ws_size; (void)in_sizes; (void)n_in; (void)out_size;

    cvt_h_kernel<<<(N_NODES * IN_F + 255) / 256, 256, 0, stream>>>(h_f, hb, N_NODES * IN_F);
    cvt_WT_kernel<<<(HEADS * IN_F * OUT_F + 255) / 256, 256, 0, stream>>>(W_f, WT);

    // 8 heads * 256 n-tiles * 4 o-tiles = 8192 waves, 8 waves/block
    wh_gemm_kernel<<<(HEADS * (N_NODES / 16) * (OUT_F / 16)) / 8, 256, 0, stream>>>(hb, WT, WhT);

    src_dst_kernel<<<(HEADS * N_NODES + 255) / 256, 256, 0, stream>>>(WhT, a_f, srcv, dstv);
    dmax_kernel<<<HEADS, 256, 0, stream>>>(dstv, dmx);

    // 256 i-tiles, 8 waves/block (one head per wave)
    gat_attn_kernel<<<N_NODES / 16, 256, 0, stream>>>(adj_f, WhT, srcv, dstv, dmx, out_f);
}
